// Conv2d_StridesAsInput_4655744548976
// MI455X (gfx1250) — matrix operand order repaired and checked by reading, not compiled, not run on hardware
//
#include <hip/hip_runtime.h>

// ---------------------------------------------------------------------------
// Fractional-stride conv (stride=2.5, pad=1, dil=1, 3x3) as im2col + f16 WMMA
// GEMM with fp32 accumulate on gfx1250.
//   x:[32,128,64,64] f32  weight:[256,128,3,3] f32  bias:[256] f32
//   out:[32,256,32,32] f32
// GEMM: M = 32768, N = 256, K = 1152.  Async global->LDS 3-stage pipeline,
// stage indices fully constant-folded (loop grouped by pipeline depth).
// ---------------------------------------------------------------------------

typedef __attribute__((ext_vector_type(16))) _Float16 v16h;
typedef __attribute__((ext_vector_type(8)))  float    v8f;

#define C_IN   128
#define H_IN   64
#define W_IN   64
#define N_OUT  256
#define K_TOT  1152            // 128 * 9
#define M_TOT  32768           // 32 * 32 * 32

#define BM     128
#define BN     128
#define BK     32
#define LDAH   (BK + 8)        // LDS row stride in halves (80 B, bank-conflict free)
#define NKIT   (K_TOT / BK)    // 36  (divisible by NSTG)
#define NSTG   3               // LDS pipeline depth

union Frag16 { v16h v; uint4 q[2]; };

// Two async global->LDS b128 copies: {base, base+16B}. INST_OFFSET is added to
// BOTH the LDS and global addresses, and our LDS/global deltas match (16 B).
__device__ __forceinline__ void async_copy_32B(unsigned lds, const _Float16* g)
{
    asm volatile("global_load_async_to_lds_b128 %0, %1, off\n\t"
                 "global_load_async_to_lds_b128 %0, %1, off offset:16"
                 :: "v"(lds), "v"(g) : "memory");
}

// -------------------------- im2col (bilinear) ------------------------------
__global__ void __launch_bounds__(256)
im2col_kernel(const float* __restrict__ x,
              const float* __restrict__ stride_h,
              const float* __restrict__ stride_w,
              _Float16* __restrict__ A)
{
    const int idx = blockIdx.x * 256 + threadIdx.x;   // exact: 2^22 threads
    const int ow = idx & 31;
    const int oh = (idx >> 5) & 31;
    const int c  = (idx >> 10) & 127;
    const int b  = idx >> 17;

    const float sh = stride_h[0];
    const float sw = stride_w[0];
    const float ph = (float)oh * sh - 1.0f;
    const float pw = (float)ow * sw - 1.0f;
    const int   h0 = (int)floorf(ph);
    const int   w0 = (int)floorf(pw);
    const float fh = ph - (float)h0;
    const float fw = pw - (float)w0;

    const float* __restrict__ xp = x + (size_t)((b * C_IN + c) * H_IN) * W_IN;

    float t[4][4];
#pragma unroll
    for (int r = 0; r < 4; ++r) {
        const int  hh = h0 + r;
        const bool hv = (hh >= 0) & (hh < H_IN);
#pragma unroll
        for (int s = 0; s < 4; ++s) {
            const int  wi = w0 + s;
            const bool wv = (wi >= 0) & (wi < W_IN);
            t[r][s] = (hv & wv) ? xp[hh * W_IN + wi] : 0.0f;
        }
    }

    const int m = (b << 10) + (oh << 5) + ow;
    _Float16* __restrict__ ap = A + (size_t)m * K_TOT + c * 9;
#pragma unroll
    for (int kh = 0; kh < 3; ++kh) {
#pragma unroll
        for (int kw = 0; kw < 3; ++kw) {
            const float v0 = t[kh][kw]   + fw * (t[kh][kw + 1]     - t[kh][kw]);
            const float v1 = t[kh+1][kw] + fw * (t[kh + 1][kw + 1] - t[kh+1][kw]);
            ap[kh * 3 + kw] = (_Float16)(v0 + fh * (v1 - v0));
        }
    }
}

// -------------------- weight f32 -> f16 (already [N][K]) -------------------
__global__ void __launch_bounds__(256)
wcvt_kernel(const float* __restrict__ w, _Float16* __restrict__ bh)
{
    const int i = blockIdx.x * 256 + threadIdx.x;     // exact: 294912 threads
    bh[i] = (_Float16)w[i];
}

// ------------------------------- WMMA GEMM ---------------------------------
// 8 wave32s in a 2(M) x 4(N) grid, each wave a 64x32 tile (4x2 accumulators).
// 3-stage async LDS pipeline with compile-time stage indices.
__global__ void __launch_bounds__(256)
gemm_kernel(const _Float16* __restrict__ A,
            const _Float16* __restrict__ Bm,
            const float* __restrict__ bias,
            float* __restrict__ out)
{
    __shared__ __align__(16) _Float16 sA[NSTG][BM * LDAH];
    __shared__ __align__(16) _Float16 sB[NSTG][BN * LDAH];

    const int tid    = threadIdx.x;
    const int lane   = tid & 31;
    const int wid    = tid >> 5;
    const int waveM  = wid & 1;
    const int waveN  = wid >> 1;
    const int laneLo = lane & 15;
    const int laneHi = lane >> 4;

    const int mBlock = blockIdx.x * BM;   // 256 blocks in M
    const int nBlock = blockIdx.y * BN;   // 2 blocks in N

    // Staging: each thread moves 32B of A and 32B of B per K-step.
    const int ldRow = tid >> 1;           // 0..127
    const int ldCol = (tid & 1) * 16;     // 0 or 16 halves

    const _Float16* gA = A  + (size_t)(mBlock + ldRow) * K_TOT + ldCol;
    const _Float16* gB = Bm + (size_t)(nBlock + ldRow) * K_TOT + ldCol;

    // Stage-0 LDS addresses (low 32 bits of the shared-aperture pointer);
    // other stages are constant byte offsets from these.
    const unsigned lA0 = (unsigned)(size_t)&sA[0][ldRow * LDAH + ldCol];
    const unsigned lB0 = (unsigned)(size_t)&sB[0][ldRow * LDAH + ldCol];
    const unsigned stgA = (unsigned)(BM * LDAH * sizeof(_Float16));  // bytes/stage
    const unsigned stgB = (unsigned)(BN * LDAH * sizeof(_Float16));

    // Prologue: stage tile 0.
    async_copy_32B(lA0, gA);
    async_copy_32B(lB0, gB);

    v8f acc[4][2] = {};

    for (int kt = 0; kt < NKIT; kt += NSTG) {
#pragma unroll
        for (int s = 0; s < NSTG; ++s) {          // s, nxt: compile-time consts
            const int kk  = kt + s;
            const int nxt = (s + 1 == NSTG) ? 0 : s + 1;

            if (kk + 1 < NKIT) {
                // Stream tile kk+1 while tile kk computes; async loads retire
                // in order, so asynccnt<=4 means tile kk's 4 loads landed.
                async_copy_32B(lA0 + nxt * stgA, gA + (size_t)(kk + 1) * BK);
                async_copy_32B(lB0 + nxt * stgB, gB + (size_t)(kk + 1) * BK);
                asm volatile("s_wait_asynccnt 0x4" ::: "memory");
            } else {
                asm volatile("s_wait_asynccnt 0x0" ::: "memory");
            }
            __syncthreads();   // single barrier/iter: safe with 3-stage rotation

            // A fragments (ISA 7.12.2 16-bit A 16x32): lane row M = laneLo,
            // K = laneHi*8 + {0..7} and 16 + laneHi*8 + {0..7}.
            Frag16 af[4];
#pragma unroll
            for (int mf = 0; mf < 4; ++mf) {
                const _Float16* p =
                    &sA[s][(waveM * 64 + mf * 16 + laneLo) * LDAH + laneHi * 8];
                af[mf].q[0] = *(const uint4*)(p);
                af[mf].q[1] = *(const uint4*)(p + 16);
            }
            // B fragments: lane col N = laneLo, K = laneHi*8 + {0..7} and 16 + laneHi*8 + {0..7}.
            Frag16 bf[2];
#pragma unroll
            for (int nf = 0; nf < 2; ++nf) {
                const _Float16* p =
                    &sB[s][(waveN * 32 + nf * 16 + laneLo) * LDAH + laneHi * 8];
                bf[nf].q[0] = *(const uint4*)(p);
                bf[nf].q[1] = *(const uint4*)(p + 16);
            }

#pragma unroll
            for (int mf = 0; mf < 4; ++mf)
#pragma unroll
                for (int nf = 0; nf < 2; ++nf)
                    acc[mf][nf] = __builtin_amdgcn_wmma_f32_16x16x32_f16(
                        false, af[mf].v, false, bf[nf].v,
                        (short)0, acc[mf][nf], false, false);
        }
    }

    // Epilogue: element v -> M = laneHi*8 + v (8 consecutive m, same batch),
    // N = laneLo. out[b][n][oh][ow], m = b*1024 + oh*32 + ow => two float4
    // stores per fragment.
#pragma unroll
    for (int nf = 0; nf < 2; ++nf) {
        const int   n  = nBlock + waveN * 32 + nf * 16 + laneLo;
        const float bv = bias[n];
#pragma unroll
        for (int mf = 0; mf < 4; ++mf) {
            const int m0 = mBlock + waveM * 64 + mf * 16 + laneHi * 8;
            const int b  = m0 >> 10;
            float* po = out + (size_t)b * (N_OUT * 1024) + (size_t)n * 1024
                            + (m0 & 1023);
            const v8f a = acc[mf][nf];
            float4 r0 = make_float4(a[0] + bv, a[1] + bv, a[2] + bv, a[3] + bv);
            float4 r1 = make_float4(a[4] + bv, a[5] + bv, a[6] + bv, a[7] + bv);
            *(float4*)(po)     = r0;
            *(float4*)(po + 4) = r1;
        }
    }
}

// ------------------------------- launcher ----------------------------------
extern "C" void kernel_launch(void* const* d_in, const int* in_sizes, int n_in,
                              void* d_out, int out_size, void* d_ws, size_t ws_size,
                              hipStream_t stream)
{
    const float* x    = (const float*)d_in[0];
    const float* w    = (const float*)d_in[1];
    const float* bias = (const float*)d_in[2];
    const float* sh   = (const float*)d_in[3];
    const float* sw   = (const float*)d_in[4];

    const size_t A_BYTES = (size_t)M_TOT * K_TOT * sizeof(_Float16);  // 75,497,472
    _Float16* Acol = (_Float16*)d_ws;
    _Float16* Bcol = (_Float16*)((char*)d_ws + A_BYTES);

    im2col_kernel<<<16384, 256, 0, stream>>>(x, sh, sw, Acol);
    wcvt_kernel<<<1152, 256, 0, stream>>>(w, Bcol);
    gemm_kernel<<<dim3(256, 2, 1), 256, 0, stream>>>(Acol, Bcol, bias, (float*)d_out);
}
